// NCF_26972394619447
// MI455X (gfx1250) — matrix operand order repaired and checked by reading, not compiled, not run on hardware
//
#include <hip/hip_runtime.h>
#include <hip/hip_bf16.h>

typedef _Float16 half_t;
typedef __attribute__((ext_vector_type(16))) _Float16 v16h;
typedef __attribute__((ext_vector_type(8)))  _Float16 v8h;
typedef __attribute__((ext_vector_type(2)))  _Float16 v2h;
typedef __attribute__((ext_vector_type(8)))  float    v8f;
typedef __attribute__((ext_vector_type(4)))  float    v4f;
typedef __attribute__((ext_vector_type(2)))  float    v2f;

#define WAVES 4
#define E 64          // EMBED
#define TM 16         // rows per wave tile

// In-wave LDS fence: DS ops are in-order per wave; this stops the compiler
// from reordering cross-lane LDS traffic and drains the DS counter.
__device__ __forceinline__ void lds_fence() {
    asm volatile("s_wait_dscnt 0" ::: "memory");
}

__device__ __forceinline__ v16h cat8(v8h lo, v8h hi) {
    v16h r;
#pragma unroll
    for (int i = 0; i < 8; ++i) { r[i] = lo[i]; r[i + 8] = hi[i]; }
    return r;
}

// A-fragment (16x32 f16, MxK) per ISA layout:
// lane m (0..15): K = k0+0..7 and k0+16..23 ; lane 16+m: K = k0+8..15 and k0+24..31
__device__ __forceinline__ v16h frag_a(const half_t* buf, int stride, int lane, int k0) {
    int m = lane & 15, h = lane >> 4;
    const half_t* p = buf + m * stride + k0 + h * 8;
    return cat8(*(const v8h*)p, *(const v8h*)(p + 16));
}

// B-fragment (32x16 f16, KxN) from transposed weights wt[n][k]:
// lane n (0..15): K = k0+0..7 and k0+16..23 ; lane 16+n: K = k0+8..15 and k0+24..31
__device__ __forceinline__ v16h frag_b(const half_t* wt, int stride, int lane, int n0, int k0) {
    int n = lane & 15, kb = (lane >> 4) * 8;
    const half_t* p = wt + (n0 + n) * stride + k0 + kb;
    return cat8(*(const v8h*)p, *(const v8h*)(p + 16));
}

__device__ __forceinline__ v8f wmma32(v16h a, v16h b, v8f c) {
    return __builtin_amdgcn_wmma_f32_16x16x32_f16(false, a, false, b, (short)0, c, false, false);
}

// Store a 16x16 f32 accumulator tile (C/D layout: lane n=L&15, VGPR j -> row j+8*(L>>4))
// with (register) bias + ReLU, converted to f16.
__device__ __forceinline__ void store_tile(half_t* dst, int stride, int lane, int n0,
                                           v8f acc, float bv) {
    int n = lane & 15, h = lane >> 4;
#pragma unroll
    for (int j = 0; j < 8; ++j) {
        float v = acc[j] + bv;
        v = v > 0.f ? v : 0.f;
        dst[(j + 8 * h) * stride + n0 + n] = (half_t)v;
    }
}

__global__ __launch_bounds__(WAVES * 32)
void ncf_fused_kernel(const int* __restrict__ user, const int* __restrict__ item,
                      const float* __restrict__ Wu_gmf, const float* __restrict__ bu_gmf,
                      const float* __restrict__ Wu_mlp, const float* __restrict__ bu_mlp,
                      const float* __restrict__ Wi_gmf, const float* __restrict__ bi_gmf,
                      const float* __restrict__ Wi_mlp, const float* __restrict__ bi_mlp,
                      const float* __restrict__ W1, const float* __restrict__ b1,
                      const float* __restrict__ W2, const float* __restrict__ b2,
                      const float* __restrict__ W3, const float* __restrict__ b3,
                      const float* __restrict__ Wp, const float* __restrict__ bp,
                      float* __restrict__ out, int num_total, int nrows) {
    // Transposed f16 weights + head weights, shared by the whole block.
    __shared__ half_t w1t[128 * 128];   // W1^T : [out=128][in=128]
    __shared__ half_t w2t[64 * 128];    // W2^T : [out=64 ][in=128]
    __shared__ half_t w3t[32 * 64];     // W3^T : [out=32 ][in=64 ]
    __shared__ float  wp_lds[96];       // Wp
    // Per-wave activation buffers.
    __shared__ half_t sbufA[WAVES][TM * 128];
    __shared__ half_t sbufB[WAVES][TM * 128];
    __shared__ float  sgmf[WAVES][TM * E];
    __shared__ int    sidx[WAVES][2 * TM];   // [0..15]=user idx, [16..31]=item idx

    const int tid  = threadIdx.x;
    const int wave = tid >> 5;
    const int lane = tid & 31;

    // ---- stage transposed f16 weights + Wp into LDS ----
    for (int i = tid; i < 128 * 128; i += blockDim.x) {
        int k = i >> 7, n = i & 127;
        w1t[n * 128 + k] = (half_t)W1[i];
    }
    for (int i = tid; i < 128 * 64; i += blockDim.x) {
        int k = i >> 6, n = i & 63;
        w2t[n * 128 + k] = (half_t)W2[i];
    }
    for (int i = tid; i < 64 * 32; i += blockDim.x) {
        int k = i >> 5, n = i & 31;
        w3t[n * 64 + k] = (half_t)W3[i];
    }
    for (int i = tid; i < 96; i += blockDim.x) wp_lds[i] = Wp[i];
    __syncthreads();

    // ---- hoist all loop-invariant biases into registers ----
    const v2f bum = ((const v2f*)bu_mlp)[lane];    // cols 2*lane, 2*lane+1
    const v2f bim = ((const v2f*)bi_mlp)[lane];
    const v2f bug = ((const v2f*)bu_gmf)[lane];
    const v2f big = ((const v2f*)bi_gmf)[lane];
    float b1v[8], b2v[4], b3v[2];
    {
        const int n = lane & 15;
#pragma unroll
        for (int nt = 0; nt < 8; ++nt) b1v[nt] = b1[nt * 16 + n];
#pragma unroll
        for (int nt = 0; nt < 4; ++nt) b2v[nt] = b2[nt * 16 + n];
#pragma unroll
        for (int nt = 0; nt < 2; ++nt) b3v[nt] = b3[nt * 16 + n];
    }
    const float bpv = bp[0];

    half_t* bufA = sbufA[wave];
    half_t* bufB = sbufB[wave];
    float*  gmfb = sgmf[wave];
    int*    idxb = sidx[wave];

    const int tiles  = (nrows + TM - 1) / TM;
    const int gwave  = blockIdx.x * WAVES + wave;
    const int nwaves = gridDim.x * WAVES;

    for (int t = gwave; t < tiles; t += nwaves) {
        const int r0 = t * TM;

        // ---- phase 0: row indices -> (user, item) ids (prefetched last iter) ----
        if (lane < TM) {
            int gr = r0 + lane;
            if (gr >= nrows) gr = nrows - 1;
            int b = gr / num_total;
            int n = gr - b * num_total;
            idxb[lane]      = user[b];
            idxb[TM + lane] = item[b * num_total + n];
        }
        lds_fence();

        // ---- phase 1: gather embeddings -> inter (f16) and gmf (f32) ----
        // Each lane handles cols {2*lane, 2*lane+1} of every row (float2 loads).
#pragma unroll 4
        for (int row = 0; row < TM; ++row) {
            int u  = idxb[row];
            int it = idxb[TM + row];
            v2f eu = ((const v2f*)(Wu_mlp + (size_t)u  * E))[lane] + bum;
            v2f ei = ((const v2f*)(Wi_mlp + (size_t)it * E))[lane] + bim;
            v2f gu = ((const v2f*)(Wu_gmf + (size_t)u  * E))[lane] + bug;
            v2f gi = ((const v2f*)(Wi_gmf + (size_t)it * E))[lane] + big;
            v2h hu; hu[0] = (half_t)eu[0]; hu[1] = (half_t)eu[1];
            v2h hi; hi[0] = (half_t)ei[0]; hi[1] = (half_t)ei[1];
            *(v2h*)(bufA + row * 128 +     2 * lane) = hu;
            *(v2h*)(bufA + row * 128 + E + 2 * lane) = hi;
            *(v2f*)(gmfb + row * E + 2 * lane) = gu * gi;
        }
        lds_fence();

        // ---- layer 1: [16x128] @ [128x128] -> h1 in bufB ----
        {
            v16h a0 = frag_a(bufA, 128, lane, 0);
            v16h a1 = frag_a(bufA, 128, lane, 32);
            v16h a2 = frag_a(bufA, 128, lane, 64);
            v16h a3 = frag_a(bufA, 128, lane, 96);
#pragma unroll
            for (int nt = 0; nt < 8; ++nt) {
                v8f acc = {};
                acc = wmma32(a0, frag_b(w1t, 128, lane, nt * 16, 0),  acc);
                acc = wmma32(a1, frag_b(w1t, 128, lane, nt * 16, 32), acc);
                acc = wmma32(a2, frag_b(w1t, 128, lane, nt * 16, 64), acc);
                acc = wmma32(a3, frag_b(w1t, 128, lane, nt * 16, 96), acc);
                store_tile(bufB, 128, lane, nt * 16, acc, b1v[nt]);
            }
        }
        lds_fence();

        // ---- prefetch next tile's indices + embedding rows (hides gather latency) ----
        {
            int tn = t + nwaves;
            if (tn < tiles && lane < TM) {
                int gr = tn * TM + lane;
                if (gr >= nrows) gr = nrows - 1;
                int b = gr / num_total;
                int n = gr - b * num_total;
                int u2  = user[b];
                int it2 = item[b * num_total + n];
                const float* p;
                p = Wu_mlp + (size_t)u2  * E; __builtin_prefetch(p, 0, 3); __builtin_prefetch(p + 32, 0, 3);
                p = Wu_gmf + (size_t)u2  * E; __builtin_prefetch(p, 0, 3); __builtin_prefetch(p + 32, 0, 3);
                p = Wi_mlp + (size_t)it2 * E; __builtin_prefetch(p, 0, 3); __builtin_prefetch(p + 32, 0, 3);
                p = Wi_gmf + (size_t)it2 * E; __builtin_prefetch(p, 0, 3); __builtin_prefetch(p + 32, 0, 3);
            }
        }

        // ---- layer 2: [16x128] @ [128x64] -> h2 in bufA (stride 64) ----
        {
            v16h a0 = frag_a(bufB, 128, lane, 0);
            v16h a1 = frag_a(bufB, 128, lane, 32);
            v16h a2 = frag_a(bufB, 128, lane, 64);
            v16h a3 = frag_a(bufB, 128, lane, 96);
#pragma unroll
            for (int nt = 0; nt < 4; ++nt) {
                v8f acc = {};
                acc = wmma32(a0, frag_b(w2t, 128, lane, nt * 16, 0),  acc);
                acc = wmma32(a1, frag_b(w2t, 128, lane, nt * 16, 32), acc);
                acc = wmma32(a2, frag_b(w2t, 128, lane, nt * 16, 64), acc);
                acc = wmma32(a3, frag_b(w2t, 128, lane, nt * 16, 96), acc);
                store_tile(bufA, 64, lane, nt * 16, acc, b2v[nt]);
            }
        }
        lds_fence();

        // ---- layer 3: [16x64] @ [64x32] -> h3 in bufB (stride 32) ----
        {
            v16h a0 = frag_a(bufA, 64, lane, 0);
            v16h a1 = frag_a(bufA, 64, lane, 32);
#pragma unroll
            for (int nt = 0; nt < 2; ++nt) {
                v8f acc = {};
                acc = wmma32(a0, frag_b(w3t, 64, lane, nt * 16, 0),  acc);
                acc = wmma32(a1, frag_b(w3t, 64, lane, nt * 16, 32), acc);
                store_tile(bufB, 32, lane, nt * 16, acc, b3v[nt]);
            }
        }
        lds_fence();

        // ---- head: sigmoid( [gmf | h3] @ Wp + bp ), vectorized LDS reads ----
        if (lane < TM) {
            int gr = r0 + lane;
            if (gr < nrows) {
                float s = bpv;
                const v4f* g4 = (const v4f*)(gmfb + lane * E);
                const v4f* w4 = (const v4f*)wp_lds;
#pragma unroll
                for (int c = 0; c < E / 4; ++c) {
                    v4f a = g4[c], w = w4[c];
                    s += a[0] * w[0] + a[1] * w[1] + a[2] * w[2] + a[3] * w[3];
                }
                const v8h* h8 = (const v8h*)(bufB + lane * 32);
#pragma unroll
                for (int c = 0; c < 4; ++c) {
                    v8h hv = h8[c];
#pragma unroll
                    for (int j = 0; j < 8; ++j) s += (float)hv[j] * wp_lds[E + c * 8 + j];
                }
                out[gr] = 1.0f / (1.0f + __expf(-s));
            }
        }
        lds_fence();
    }
}

extern "C" void kernel_launch(void* const* d_in, const int* in_sizes, int n_in,
                              void* d_out, int out_size, void* d_ws, size_t ws_size,
                              hipStream_t stream) {
    const int*   user   = (const int*)  d_in[0];
    const int*   item   = (const int*)  d_in[1];
    // d_in[2] = num_total scalar (device); recompute from sizes instead.
    const float* Wu_gmf = (const float*)d_in[3];
    const float* bu_gmf = (const float*)d_in[4];
    const float* Wu_mlp = (const float*)d_in[5];
    const float* bu_mlp = (const float*)d_in[6];
    const float* Wi_gmf = (const float*)d_in[7];
    const float* bi_gmf = (const float*)d_in[8];
    const float* Wi_mlp = (const float*)d_in[9];
    const float* bi_mlp = (const float*)d_in[10];
    const float* W1     = (const float*)d_in[11];
    const float* b1     = (const float*)d_in[12];
    const float* W2     = (const float*)d_in[13];
    const float* b2     = (const float*)d_in[14];
    const float* W3     = (const float*)d_in[15];
    const float* b3     = (const float*)d_in[16];
    const float* Wp     = (const float*)d_in[17];
    const float* bp     = (const float*)d_in[18];
    float* out = (float*)d_out;

    const int batch     = in_sizes[0];                 // 4096
    const int num_total = in_sizes[1] / batch;         // 100
    const int nrows     = out_size;                    // batch * num_total

    const int blocks = 512;
    ncf_fused_kernel<<<blocks, WAVES * 32, 0, stream>>>(
        user, item, Wu_gmf, bu_gmf, Wu_mlp, bu_mlp, Wi_gmf, bi_gmf, Wi_mlp, bi_mlp,
        W1, b1, W2, b2, W3, b3, Wp, bp, out, num_total, nrows);
}
